// Attention_83837761618671
// MI455X (gfx1250) — compile-verified, matrix operand order checked by reading, not run on hardware
//
#include <hip/hip_runtime.h>
#include <hip/hip_bf16.h>

#define NHEAD 8
#define HS    64
#define SEQ   4096
#define DMODEL 512
#define DQKV  1536

typedef __attribute__((ext_vector_type(16))) _Float16 v16h;
typedef __attribute__((ext_vector_type(8)))  _Float16 v8h;
typedef __attribute__((ext_vector_type(4)))  _Float16 v4h;
typedef __attribute__((ext_vector_type(8)))  float    v8f;

// ---- WMMA helpers -----------------------------------------------------------

static __device__ __forceinline__ v8f wmma16(v16h a, v16h b, v8f c) {
  // D = A x B + C, 16x16x32 fp16 -> f32 accum
  return __builtin_amdgcn_wmma_f32_16x16x32_f16(
      /*neg_a=*/false, a, /*neg_b=*/false, b,
      /*c_mod=*/(short)0, c, /*reuse_a=*/false, /*reuse_b=*/false);
}

// Load a 16x32 fp16 fragment from a row-major matrix (contraction dim
// contiguous).  Per ISA layout: lanes 0-15 hold row (l&15) K={k0..k0+7,
// k0+16..k0+23}; lanes 16-31 hold K={k0+8..k0+15, k0+24..k0+31}.
// Works for both A and B operands of an NT GEMM (B rows == columns of B).
static __device__ __forceinline__ v16h load_frag_nt(const _Float16* __restrict__ base,
                                                    int ld, int r0, int k0) {
  const int lane = threadIdx.x & 31;
  const _Float16* p = base + (size_t)(r0 + (lane & 15)) * ld + k0 + ((lane >> 4) << 3);
  v8h lo = *(const v8h*)p;          // K = k0+kh .. +7
  v8h hi = *(const v8h*)(p + 16);   // K = k0+16+kh .. +7
  v16h r;
#pragma unroll
  for (int i = 0; i < 8; ++i) { r[i] = lo[i]; r[i + 8] = hi[i]; }
  return r;
}

// ---- Elementwise fp32 -> fp16 convert ---------------------------------------

__global__ void cvt_f32_f16(const float* __restrict__ s, _Float16* __restrict__ d, int n) {
  int i = (blockIdx.x * blockDim.x + threadIdx.x) * 4;
  if (i + 3 < n) {
    float4 v = *(const float4*)(s + i);
    v4h o;
    o[0] = (_Float16)v.x; o[1] = (_Float16)v.y;
    o[2] = (_Float16)v.z; o[3] = (_Float16)v.w;
    *(v4h*)(d + i) = o;
  }
}

// ---- Generic NT GEMM: C[M,N] = A[M,K] * B[N,K]^T ----------------------------
// 256 threads = 8 waves.  Wave tile 32x32 (2x2 WMMA accs); block tile 128x64.
// This tile size gives the compiler enough registers to software-pipeline:
// clause-grouped b128 loads a full k-step ahead with staged s_wait_loadcnt.

__global__ __launch_bounds__(256)
void gemm_nt_f16out(const _Float16* __restrict__ A, const _Float16* __restrict__ B,
                    _Float16* __restrict__ C, int lda, int ldb, int ldc, int K) {
  const int w = threadIdx.x >> 5;
  const int m0 = blockIdx.x * 128 + (w & 3) * 32;
  const int n0 = blockIdx.y * 64 + (w >> 2) * 32;
  v8f acc[2][2] = {};
  for (int k0 = 0; k0 < K; k0 += 32) {
    v16h a0 = load_frag_nt(A, lda, m0,      k0);
    v16h a1 = load_frag_nt(A, lda, m0 + 16, k0);
    v16h b0 = load_frag_nt(B, ldb, n0,      k0);
    v16h b1 = load_frag_nt(B, ldb, n0 + 16, k0);
    acc[0][0] = wmma16(a0, b0, acc[0][0]);
    acc[0][1] = wmma16(a0, b1, acc[0][1]);
    acc[1][0] = wmma16(a1, b0, acc[1][0]);
    acc[1][1] = wmma16(a1, b1, acc[1][1]);
  }
  const int lane = threadIdx.x & 31;
  const int nn = lane & 15;
  const int mb = (lane >> 4) << 3;   // C layout: VGPR r -> M = r + 8*(lane>=16)
#pragma unroll
  for (int i = 0; i < 2; ++i)
#pragma unroll
    for (int j = 0; j < 2; ++j)
#pragma unroll
      for (int r = 0; r < 8; ++r)
        C[(size_t)(m0 + 16 * i + mb + r) * ldc + n0 + 16 * j + nn] =
            (_Float16)acc[i][j][r];
}

__global__ __launch_bounds__(256)
void gemm_nt_f32out_bias(const _Float16* __restrict__ A, const _Float16* __restrict__ B,
                         const float* __restrict__ bias, float* __restrict__ C,
                         int lda, int ldb, int ldc, int K) {
  const int w = threadIdx.x >> 5;
  const int m0 = blockIdx.x * 128 + (w & 3) * 32;
  const int n0 = blockIdx.y * 64 + (w >> 2) * 32;
  v8f acc[2][2] = {};
  for (int k0 = 0; k0 < K; k0 += 32) {
    v16h a0 = load_frag_nt(A, lda, m0,      k0);
    v16h a1 = load_frag_nt(A, lda, m0 + 16, k0);
    v16h b0 = load_frag_nt(B, ldb, n0,      k0);
    v16h b1 = load_frag_nt(B, ldb, n0 + 16, k0);
    acc[0][0] = wmma16(a0, b0, acc[0][0]);
    acc[0][1] = wmma16(a0, b1, acc[0][1]);
    acc[1][0] = wmma16(a1, b0, acc[1][0]);
    acc[1][1] = wmma16(a1, b1, acc[1][1]);
  }
  const int lane = threadIdx.x & 31;
  const int nn = lane & 15;
  const int mb = (lane >> 4) << 3;
#pragma unroll
  for (int i = 0; i < 2; ++i)
#pragma unroll
    for (int j = 0; j < 2; ++j) {
      const float bv = bias[n0 + 16 * j + nn];
#pragma unroll
      for (int r = 0; r < 8; ++r)
        C[(size_t)(m0 + 16 * i + mb + r) * ldc + n0 + 16 * j + nn] =
            acc[i][j][r] + bv;
    }
}

// ---- V transpose: Vt[h*64+c][n] = qkv[n][1024 + h*64 + c] -------------------

__global__ void transposeV(const _Float16* __restrict__ qkv, _Float16* __restrict__ Vt) {
  int i = blockIdx.x * blockDim.x + threadIdx.x;   // over 512*4096
  int c = i >> 12;          // 0..511
  int n = i & 4095;
  Vt[(size_t)c * SEQ + n] = qkv[(size_t)n * DQKV + 2 * DMODEL + c];
}

// ---- Fused attention: scores -> softmax (in 256KB LDS) -> att store -> att@V

__global__ __launch_bounds__(256)
void attn_fused(const _Float16* __restrict__ qkv, const _Float16* __restrict__ Vt,
                float* __restrict__ att, _Float16* __restrict__ out1) {
  extern __shared__ float sm[];
  float* sS    = sm;                   // 16 x 4096 f32 score strip (256 KB)
  float* red   = sm + 16 * SEQ;        // 256 partials
  float* rstat = red + 256;            // 16 row stats (max, then 1/sum)
  float* pscr  = rstat + 16;           // 4 x 256 f32 partial C tiles (4 KB)

  const int h  = blockIdx.y;
  const int q0 = blockIdx.x * 16;
  const int t  = threadIdx.x;
  const int w  = t >> 5;
  const int lane = t & 31;
  const int nn = lane & 15;
  const int mb = (lane >> 4) << 3;

  const _Float16* Qh = qkv + h * HS;            // rows: token n, ld = 1536
  const _Float16* Kh = qkv + DMODEL + h * HS;

  // ---- Phase 1: S = scale * Q_tile @ K^T, whole 16x4096 strip into LDS ----
  v16h aq0 = load_frag_nt(Qh, DQKV, q0, 0);
  v16h aq1 = load_frag_nt(Qh, DQKV, q0, 32);
  const float scale = 0.125f;                   // 1/sqrt(64)
  for (int j = w; j < SEQ / 16; j += 8) {
    v16h b0 = load_frag_nt(Kh, DQKV, j * 16, 0);
    v16h b1 = load_frag_nt(Kh, DQKV, j * 16, 32);
    v8f acc = {};
    acc = wmma16(aq0, b0, acc);
    acc = wmma16(aq1, b1, acc);
#pragma unroll
    for (int r = 0; r < 8; ++r)
      sS[(mb + r) * SEQ + j * 16 + nn] = acc[r] * scale;
  }
  __syncthreads();

  // ---- Phase 2: row softmax, 16 threads per row, 256 elems per thread -----
  const int row = t >> 4, seg = t & 15;
  float* sp = sS + row * SEQ + seg * 256;

  float mx = -3.0e38f;
  for (int i = 0; i < 256; i += 4) {
    float4 v = *(const float4*)(sp + i);
    mx = fmaxf(mx, fmaxf(fmaxf(v.x, v.y), fmaxf(v.z, v.w)));
  }
  red[t] = mx;
  __syncthreads();
  if (seg == 0) {
    float m2 = red[row * 16];
    for (int i = 1; i < 16; ++i) m2 = fmaxf(m2, red[row * 16 + i]);
    rstat[row] = m2;
  }
  __syncthreads();
  const float rmax = rstat[row];

  float sum = 0.f;
  for (int i = 0; i < 256; i += 4) {
    float4 v = *(const float4*)(sp + i);
    v.x = __expf(v.x - rmax); v.y = __expf(v.y - rmax);
    v.z = __expf(v.z - rmax); v.w = __expf(v.w - rmax);
    sum += (v.x + v.y) + (v.z + v.w);
    *(float4*)(sp + i) = v;
  }
  red[t] = sum;
  __syncthreads();
  if (seg == 0) {
    float s2 = 0.f;
    for (int i = 0; i < 16; ++i) s2 += red[row * 16 + i];
    rstat[row] = 1.0f / s2;
  }
  __syncthreads();
  const float rinv = rstat[row];

  // normalize in LDS and stream att to HBM exactly once
  float* gp = att + ((size_t)h * SEQ + q0 + row) * SEQ + seg * 256;
  for (int i = 0; i < 256; i += 4) {
    float4 v = *(const float4*)(sp + i);
    v.x *= rinv; v.y *= rinv; v.z *= rinv; v.w *= rinv;
    *(float4*)(sp + i) = v;
    *(float4*)(gp + i) = v;
  }
  __syncthreads();

  // ---- Phase 3: out1_tile = att(LDS) @ V.  All 8 waves: wave w takes
  // column tile c0=(w&3)*16 and K-half (w>>2); partials pair-reduced in LDS.
  {
    const int c0 = (w & 3) * 16;
    const int kbeg = (w >> 2) * (SEQ / 2);
    const int kend = kbeg + SEQ / 2;
    const _Float16* VtH = Vt + (size_t)h * HS * SEQ;   // rows: out col c, ld = 4096
    const int m = lane & 15, kh = (lane >> 4) << 3;
    v8f acc = {};
    for (int k0 = kbeg; k0 < kend; k0 += 32) {
      const float* ap = sS + m * SEQ + k0 + kh;
      v16h a;
#pragma unroll
      for (int i = 0; i < 8; ++i) {
        a[i]     = (_Float16)ap[i];
        a[i + 8] = (_Float16)ap[i + 16];
      }
      v16h b = load_frag_nt(VtH, SEQ, c0, k0);
      acc = wmma16(a, b, acc);
    }
    if (w >= 4) {
#pragma unroll
      for (int r = 0; r < 8; ++r)
        pscr[(w - 4) * 256 + lane * 8 + r] = acc[r];
    }
    __syncthreads();
    if (w < 4) {
#pragma unroll
      for (int r = 0; r < 8; ++r)
        acc[r] += pscr[w * 256 + lane * 8 + r];
#pragma unroll
      for (int r = 0; r < 8; ++r)
        out1[(size_t)(q0 + mb + r) * DMODEL + h * HS + c0 + nn] = (_Float16)acc[r];
    }
  }
}

// ---- Host-side orchestration ------------------------------------------------

extern "C" void kernel_launch(void* const* d_in, const int* in_sizes, int n_in,
                              void* d_out, int out_size, void* d_ws, size_t ws_size,
                              hipStream_t stream) {
  (void)in_sizes; (void)n_in; (void)out_size; (void)ws_size;
  const float* x     = (const float*)d_in[0];   // [4096, 512]
  const float* Wqkv  = (const float*)d_in[1];   // [1536, 512]
  const float* Wproj = (const float*)d_in[2];   // [512, 512]
  const float* bproj = (const float*)d_in[3];   // [512]

  float* out = (float*)d_out;                       // [4096, 512]
  float* att = out + (size_t)SEQ * DMODEL;          // [8, 4096, 4096]

  // workspace carve-up (~26 MB)
  _Float16* xh   = (_Float16*)d_ws;                 // 4096 x 512
  _Float16* Wqh  = xh   + (size_t)SEQ * DMODEL;     // 1536 x 512
  _Float16* Wph  = Wqh  + (size_t)DQKV * DMODEL;    // 512 x 512
  _Float16* qkvh = Wph  + (size_t)DMODEL * DMODEL;  // 4096 x 1536
  _Float16* Vt   = qkvh + (size_t)SEQ * DQKV;       // 512 x 4096
  _Float16* o1h  = Vt   + (size_t)DMODEL * SEQ;     // 4096 x 512

  // 1) fp32 -> fp16 converts
  cvt_f32_f16<<<(SEQ * DMODEL / 4 + 255) / 256, 256, 0, stream>>>(x, xh, SEQ * DMODEL);
  cvt_f32_f16<<<(DQKV * DMODEL / 4 + 255) / 256, 256, 0, stream>>>(Wqkv, Wqh, DQKV * DMODEL);
  cvt_f32_f16<<<(DMODEL * DMODEL / 4 + 255) / 256, 256, 0, stream>>>(Wproj, Wph, DMODEL * DMODEL);

  // 2) qkv = x @ Wqkv^T   (M=4096, N=1536, K=512), block tile 128x64
  gemm_nt_f16out<<<dim3(SEQ / 128, DQKV / 64), 256, 0, stream>>>(
      xh, Wqh, qkvh, DMODEL, DMODEL, DQKV, DMODEL);

  // 3) Vt[h*64+c][n] = V[n][h*64+c]
  transposeV<<<(DMODEL * SEQ) / 256, 256, 0, stream>>>(qkvh, Vt);

  // 4) fused attention: per (16 query rows, head)
  size_t smem = (size_t)(16 * SEQ + 256 + 16 + 4 * 256) * sizeof(float);  // ~267 KB
  attn_fused<<<dim3(SEQ / 16, NHEAD), 256, smem, stream>>>(qkvh, Vt, att, o1h);

  // 5) out = out1 @ Wproj^T + bproj  (M=4096, N=512, K=512)
  gemm_nt_f32out_bias<<<dim3(SEQ / 128, DMODEL / 64), 256, 0, stream>>>(
      o1h, Wph, bproj, out, DMODEL, DMODEL, DMODEL, DMODEL);
}
